// FixedShapeConformerAttention_69432441307608
// MI455X (gfx1250) — compile-verified, matrix operand order checked by reading, not run on hardware
//
#include <hip/hip_runtime.h>
#include <hip/hip_bf16.h>
#include <math.h>

// ---------------------------------------------------------------------------
// Fixed-shape Conformer chunked attention for MI455X (gfx1250, wave32, WMMA)
//   S=8192, H=16, D=128, HD=2048, C=12, MP=12, MF=0, CTX=24, POS=13, N=683
// bf16 WMMA (v_wmma_f32_16x16x32_bf16) everywhere, fp32 accumulate.
// GEMM: 128x128 block tile, double-buffered LDS, one barrier per K-stage,
// GLOBAL_LOAD_ASYNC_TO_LDS_B128 staging (ASYNCcnt) when the toolchain
// exposes the builtin (it does here: signature is (v4i AS1*, v4i AS3*, Ii, Ii)).
// ---------------------------------------------------------------------------

typedef __bf16 bf16_t;
typedef bf16_t v16bf __attribute__((ext_vector_type(16)));
typedef bf16_t v8bf  __attribute__((ext_vector_type(8)));
typedef float  v8f   __attribute__((ext_vector_type(8)));
typedef int    v4i   __attribute__((ext_vector_type(4)));

#define SEQ   8192
#define HEADS 16
#define DH    128
#define HID   2048
#define CHK   12
#define MPAST 12
#define CTXW  24
#define POSL  13
#define NBLK  683   // ceil(SEQ / CHK); NBLK*CHK = 8196 (4 padded query rows)

#if defined(__has_builtin)
#if __has_builtin(__builtin_amdgcn_global_load_async_to_lds_b128)
#define ASYNC_LDS 1
#else
#define ASYNC_LDS 0
#endif
#else
#define ASYNC_LDS 0
#endif

// ------------------------------ WMMA helpers -------------------------------

__device__ __forceinline__ v16bf frag_zero() {
  union { v16bf v; unsigned u[8]; } z;
#pragma unroll
  for (int i = 0; i < 8; ++i) z.u[i] = 0u;
  return z.v;
}

__device__ __forceinline__ v8f acc_zero() {
  union { v8f v; float f[8]; } z;
#pragma unroll
  for (int i = 0; i < 8; ++i) z.f[i] = 0.0f;
  return z.v;
}

// One lane's share of a 16x32 bf16 A/B operand (CDNA5 ISA layout):
// K = [koff..koff+7] in VGPRs 0-3, K = [koff+16..koff+23] in VGPRs 4-7,
// koff = (lane<16) ? 0 : 8. p is already offset by koff.
__device__ __forceinline__ v16bf frag_ld(const bf16_t* p) {
  union { v16bf v; v8bf h[2]; } u;
  u.h[0] = *(const v8bf*)(p);
  u.h[1] = *(const v8bf*)(p + 16);
  return u.v;
}

__device__ __forceinline__ v8f wmma_bf(v16bf a, v16bf b, v8f c) {
  // (neg_a, A, neg_b, B, c_mod, C, reuse_a, reuse_b)
  return __builtin_amdgcn_wmma_f32_16x16x32_bf16(false, a, false, b,
                                                 (short)0, c, false, false);
}

__device__ __forceinline__ void wait_async_then_barrier() {
#if ASYNC_LDS
#if __has_builtin(__builtin_amdgcn_s_wait_asynccnt)
  __builtin_amdgcn_s_wait_asynccnt(0);
#else
  asm volatile("s_wait_asynccnt 0x0" ::: "memory");
#endif
#endif
  __syncthreads();
}

// ------------------------------ prep kernels -------------------------------

__global__ void conv_x_kernel(const float* __restrict__ x,
                              bf16_t* __restrict__ xb, long n) {
  long i = (long)blockIdx.x * blockDim.x + threadIdx.x;
  if (i < n) xb[i] = (bf16_t)x[i];
}

// W is [K=HID, N=HID] row-major fp32; store Wt[N][K] bf16 so the WMMA B
// operand loads row-major exactly like A.
__global__ void transpose_w_kernel(const float* __restrict__ W,
                                   bf16_t* __restrict__ Wt) {
  long idx = (long)blockIdx.x * blockDim.x + threadIdx.x;  // HID*HID threads
  int i = (int)(idx >> 11);        // input dim (K)
  int j = (int)(idx & (HID - 1));  // output dim (N)
  Wt[(long)j * HID + i] = (bf16_t)W[idx];
}

// position embeddings [POSL, HID] -> zero-padded [128, HID] bf16
__global__ void posemb_kernel(const float* __restrict__ pe,
                              bf16_t* __restrict__ peb) {
  int idx = blockIdx.x * blockDim.x + threadIdx.x;  // 128*HID threads
  int row = idx >> 11;
  int col = idx & (HID - 1);
  peb[idx] = (row < POSL) ? (bf16_t)pe[row * HID + col] : (bf16_t)0.0f;
}

// qscale[d] = D^-0.5 * softplus(per_dim_scale[d]) / softplus(0)
__global__ void qscale_kernel(const float* __restrict__ pds,
                              float* __restrict__ qs) {
  int d = threadIdx.x;
  if (d < DH) {
    float x = pds[d];
    float sp = (x > 15.0f) ? x : log1pf(expf(x));
    qs[d] = 0.08838834764831845f /* 128^-0.5 */ *
            1.4426950408889634f  /* 1/softplus(0) */ * sp;
  }
}

// ------------------------------- main GEMM ---------------------------------
// C[M,N] = A[M,K] @ Bt[N,K]^T, bf16 in, f32 accumulate.
// Block tile 128x128, 8 waves of 32x64 (2x4 WMMA tiles), BK=32, LDS double
// buffered (one barrier per stage). K must be a multiple of 64; M, N
// multiples of 128 (true for 8192/2048/128 here).
// OUT_MODE: 0 = fp32 out, 1 = bf16 out, 2 = bf16 out * colscale[col % 128]

struct StageRegs { v8bf a0, a1, b0, b1; };

__device__ __forceinline__ void stage_chunk_ids(int tid, int& r0, int& kc0,
                                                int& r1, int& kc1) {
  r0 = tid >> 2;            // 512 chunks of 16B per 128x32 tile, 2/thread
  kc0 = (tid & 3) << 3;
  int ch1 = tid + 256;
  r1 = ch1 >> 2;
  kc1 = (ch1 & 3) << 3;
}

__device__ __forceinline__ void stage_issue(StageRegs& s,
    const bf16_t* __restrict__ A, const bf16_t* __restrict__ Bt,
    long rowBase, long colBase, int K, int k0, int tid) {
  int r0, kc0, r1, kc1;
  stage_chunk_ids(tid, r0, kc0, r1, kc1);
  s.a0 = *(const v8bf*)&A[(rowBase + r0) * (long)K + k0 + kc0];
  s.a1 = *(const v8bf*)&A[(rowBase + r1) * (long)K + k0 + kc1];
  s.b0 = *(const v8bf*)&Bt[(colBase + r0) * (long)K + k0 + kc0];
  s.b1 = *(const v8bf*)&Bt[(colBase + r1) * (long)K + k0 + kc1];
}

__device__ __forceinline__ void stage_commit(StageRegs& s,
    bf16_t* AsBuf, bf16_t* BsBuf, int tid) {
  int r0, kc0, r1, kc1;
  stage_chunk_ids(tid, r0, kc0, r1, kc1);
  *(v8bf*)&AsBuf[r0 * 32 + kc0] = s.a0;
  *(v8bf*)&AsBuf[r1 * 32 + kc1] = s.a1;
  *(v8bf*)&BsBuf[r0 * 32 + kc0] = s.b0;
  *(v8bf*)&BsBuf[r1 * 32 + kc1] = s.b1;
}

#if ASYNC_LDS
__device__ __forceinline__ void stage_async(
    const bf16_t* __restrict__ A, const bf16_t* __restrict__ Bt,
    bf16_t* AsBuf, bf16_t* BsBuf,
    long rowBase, long colBase, int K, int k0, int tid) {
  int r0, kc0, r1, kc1;
  stage_chunk_ids(tid, r0, kc0, r1, kc1);
#define GASYNC128(g, l)                                                     \
  __builtin_amdgcn_global_load_async_to_lds_b128(                           \
      (__attribute__((address_space(1))) v4i*)(g),                          \
      (__attribute__((address_space(3))) v4i*)(l), 0, 0)
  GASYNC128(&A[(rowBase + r0) * (long)K + k0 + kc0], &AsBuf[r0 * 32 + kc0]);
  GASYNC128(&A[(rowBase + r1) * (long)K + k0 + kc1], &AsBuf[r1 * 32 + kc1]);
  GASYNC128(&Bt[(colBase + r0) * (long)K + k0 + kc0], &BsBuf[r0 * 32 + kc0]);
  GASYNC128(&Bt[(colBase + r1) * (long)K + k0 + kc1], &BsBuf[r1 * 32 + kc1]);
#undef GASYNC128
}
#endif

__device__ __forceinline__ void compute_tile(const bf16_t* AsBuf,
    const bf16_t* BsBuf, v8f acc[2][4], int wm, int wn, int ml, int koff) {
  v16bf af[2], bfr[4];
#pragma unroll
  for (int tm = 0; tm < 2; ++tm)
    af[tm] = frag_ld(&AsBuf[(wm * 32 + tm * 16 + ml) * 32 + koff]);
#pragma unroll
  for (int tn = 0; tn < 4; ++tn)
    bfr[tn] = frag_ld(&BsBuf[(wn * 64 + tn * 16 + ml) * 32 + koff]);
#pragma unroll
  for (int tm = 0; tm < 2; ++tm)
#pragma unroll
    for (int tn = 0; tn < 4; ++tn)
      acc[tm][tn] = wmma_bf(af[tm], bfr[tn], acc[tm][tn]);
}

template <int OUT_MODE>
__global__ __launch_bounds__(256) void gemm_bf16(
    const bf16_t* __restrict__ A, const bf16_t* __restrict__ Bt,
    void* __restrict__ Cout, const float* __restrict__ colscale,
    int M, int Nn, int K) {
  __shared__ alignas(16) bf16_t As[2][128 * 32];
  __shared__ alignas(16) bf16_t Bs[2][128 * 32];

  const int tid  = threadIdx.x;
  const int lane = tid & 31;
  const int wave = tid >> 5;
  const int wm   = wave & 3;   // 4 row-groups of 32
  const int wn   = wave >> 2;  // 2 col-groups of 64
  const long rowBase = (long)blockIdx.x * 128;
  const long colBase = (long)blockIdx.y * 128;
  const int koff = (lane & 16) ? 8 : 0;
  const int ml   = lane & 15;

  v8f acc[2][4];
#pragma unroll
  for (int a = 0; a < 2; ++a)
#pragma unroll
    for (int b = 0; b < 4; ++b) acc[a][b] = acc_zero();

  // ---- prologue: stage K-tile 0 into buffer 0 ----
#if ASYNC_LDS
  stage_async(A, Bt, As[0], Bs[0], rowBase, colBase, K, 0, tid);
#else
  StageRegs sr;
  stage_issue(sr, A, Bt, rowBase, colBase, K, 0, tid);
  stage_commit(sr, As[0], Bs[0], tid);
#endif
  wait_async_then_barrier();

  // ---- double-buffered main loop, K unrolled by 2 (one barrier/stage) ----
  for (int k0 = 0; k0 < K; k0 += 64) {
#if ASYNC_LDS
    stage_async(A, Bt, As[1], Bs[1], rowBase, colBase, K, k0 + 32, tid);
    compute_tile(As[0], Bs[0], acc, wm, wn, ml, koff);
    wait_async_then_barrier();
    if (k0 + 64 < K)
      stage_async(A, Bt, As[0], Bs[0], rowBase, colBase, K, k0 + 64, tid);
    compute_tile(As[1], Bs[1], acc, wm, wn, ml, koff);
    wait_async_then_barrier();
#else
    stage_issue(sr, A, Bt, rowBase, colBase, K, k0 + 32, tid);
    compute_tile(As[0], Bs[0], acc, wm, wn, ml, koff);  // hide load latency
    stage_commit(sr, As[1], Bs[1], tid);
    __syncthreads();
    if (k0 + 64 < K)
      stage_issue(sr, A, Bt, rowBase, colBase, K, k0 + 64, tid);
    compute_tile(As[1], Bs[1], acc, wm, wn, ml, koff);
    if (k0 + 64 < K) stage_commit(sr, As[0], Bs[0], tid);
    __syncthreads();
#endif
  }

  // ---- epilogue: D layout -> VGPR r holds M=r (lanes 0-15) / r+8 (16-31),
  // N = lane&15 ----
  const int hi8 = (lane & 16) ? 8 : 0;
#pragma unroll
  for (int tm = 0; tm < 2; ++tm)
#pragma unroll
    for (int tn = 0; tn < 4; ++tn) {
      union { v8f v; float f[8]; } res;
      res.v = acc[tm][tn];
#pragma unroll
      for (int r = 0; r < 8; ++r) {
        long row = rowBase + wm * 32 + tm * 16 + r + hi8;
        long col = colBase + wn * 64 + tn * 16 + ml;
        float vv = res.f[r];
        if (OUT_MODE == 0) {
          ((float*)Cout)[row * Nn + col] = vv;
        } else {
          if (OUT_MODE == 2) vv *= colscale[col & (DH - 1)];
          ((bf16_t*)Cout)[row * Nn + col] = (bf16_t)vv;
        }
      }
    }
}

// ---------------------------- attention kernel -----------------------------
// One wave per (chunk n, head h). grid = (NBLK, HEADS), block = 32.
//  scores: ac[16x32] = qb @ kb^T (2 WMMA N-tiles), bd[16x16] = qb @ relk^T
//  shift trick: score(c,j) = ac(c,j) + (0 <= j-c <= 12 ? bd(c, j-c) : 0)
//  softcap 50*tanh(z/50), softmax over j=0..23 (reference has NO key mask)
//  out[16x128] = attn(bf16) @ vb via 8 WMMA N-tiles, V transposed in LDS.

__global__ __launch_bounds__(32) void attn_kernel(
    const bf16_t* __restrict__ q, const bf16_t* __restrict__ k,
    const bf16_t* __restrict__ v, const bf16_t* __restrict__ relk,
    bf16_t* __restrict__ attn_out) {
  __shared__ alignas(16) float  sc[16 * 32];
  __shared__ alignas(16) float  sbd[16 * 16];
  __shared__ alignas(16) bf16_t attn_s[16 * 32];
  __shared__ alignas(16) bf16_t vbT[128 * 32];  // [d][j], j padded to 32

  const int n    = blockIdx.x;
  const int h    = blockIdx.y;
  const int lane = threadIdx.x;
  const int ml   = lane & 15;
  const int koff = (lane & 16) ? 8 : 0;
  const int hi8  = (lane & 16) ? 8 : 0;

  // ---- score phase: A = qb rows (lane -> m), B = kb rows (lane -> j) ----
  const int  qrow = n * CHK + ml;
  const bool qv   = (ml < CHK) && (qrow < SEQ);
  const int  kr0  = n * CHK - MPAST + ml;       // j = ml in [0,15]
  const bool kv0  = (kr0 >= 0);                 // kr0 < SEQ always
  const int  j1   = 16 + ml;
  const int  kr1  = n * CHK - MPAST + j1;       // >= 4, never negative
  const bool kv1  = (j1 < CTXW) && (kr1 < SEQ);

  const bf16_t* qp  = q    + (long)qrow * HID + h * DH + koff;
  const bf16_t* kp0 = k    + (long)kr0  * HID + h * DH + koff;
  const bf16_t* kp1 = k    + (long)kr1  * HID + h * DH + koff;
  const bf16_t* rp  = relk + (long)ml   * HID + h * DH + koff;  // rows 13..15 zero

  v8f ac0 = acc_zero(), ac1 = acc_zero(), bd = acc_zero();
#pragma unroll
  for (int kk = 0; kk < DH; kk += 32) {
    v16bf afq = qv  ? frag_ld(qp  + kk) : frag_zero();
    v16bf b0  = kv0 ? frag_ld(kp0 + kk) : frag_zero();
    v16bf b1  = kv1 ? frag_ld(kp1 + kk) : frag_zero();
    v16bf br  = frag_ld(rp + kk);
    ac0 = wmma_bf(afq, b0, ac0);
    ac1 = wmma_bf(afq, b1, ac1);
    bd  = wmma_bf(afq, br, bd);
  }
  {
    union { v8f v; float f[8]; } r0, r1, rb;
    r0.v = ac0; r1.v = ac1; rb.v = bd;
#pragma unroll
    for (int r = 0; r < 8; ++r) {
      sc[(r + hi8) * 32 + ml]      = r0.f[r];
      sc[(r + hi8) * 32 + 16 + ml] = r1.f[r];
      sbd[(r + hi8) * 16 + ml]     = rb.f[r];
    }
  }
  // zero vbT while scores land (disjoint LDS)
  {
    unsigned* vz = (unsigned*)vbT;
    for (int i = lane; i < (128 * 32) / 2; i += 32) vz[i] = 0u;
  }
  __syncthreads();

  // ---- softcap + softmax, one row per lane (lanes 0-15) ----
  if (lane < 16) {
    float t[CTXW];
    float mx = -1e30f;
#pragma unroll
    for (int j = 0; j < CTXW; ++j) {
      float s = sc[lane * 32 + j];
      int   p = j - lane;
      if (p >= 0 && p < POSL) s += sbd[lane * 16 + p];
      s = 50.0f * tanhf(s * 0.02f);
      t[j] = s;
      mx = fmaxf(mx, s);
    }
    float sum = 0.0f;
#pragma unroll
    for (int j = 0; j < CTXW; ++j) {
      float e = __expf(t[j] - mx);
      t[j] = e;
      sum += e;
    }
    float inv = 1.0f / sum;
#pragma unroll
    for (int j = 0; j < CTXW; ++j)
      attn_s[lane * 32 + j] = (bf16_t)(t[j] * inv);
#pragma unroll
    for (int j = CTXW; j < 32; ++j) attn_s[lane * 32 + j] = (bf16_t)0.0f;
  }
  __syncthreads();

  // ---- stage V transposed: vbT[d][j] = v[kb_row(j)][h*128+d] ----
  for (int i = lane; i < CTXW * DH; i += 32) {
    int j = i >> 7;          // 0..23
    int d = i & (DH - 1);    // 0..127 (consecutive lanes -> contiguous d)
    int sr = n * CHK - MPAST + j;
    if (sr >= 0 && sr < SEQ) vbT[d * 32 + j] = v[(long)sr * HID + h * DH + d];
  }
  __syncthreads();

  // ---- out = attn @ vb : 1 A-frag x 8 B-frags, K = 32 (j padded) ----
  v16bf afr = frag_ld(&attn_s[ml * 32 + koff]);
  v8f ot[8];
#pragma unroll
  for (int dt = 0; dt < 8; ++dt) {
    v16bf bfv = frag_ld(&vbT[(dt * 16 + ml) * 32 + koff]);
    ot[dt] = wmma_bf(afr, bfv, acc_zero());
  }

#pragma unroll
  for (int dt = 0; dt < 8; ++dt) {
    union { v8f v; float f[8]; } rr;
    rr.v = ot[dt];
#pragma unroll
    for (int r = 0; r < 8; ++r) {
      int c  = r + hi8;
      int sr = n * CHK + c;
      if (c < CHK && sr < SEQ)
        attn_out[(long)sr * HID + h * DH + dt * 16 + ml] = (bf16_t)rr.f[r];
    }
  }
}

// ------------------------------- launcher ----------------------------------

extern "C" void kernel_launch(void* const* d_in, const int* in_sizes, int n_in,
                              void* d_out, int out_size, void* d_ws,
                              size_t ws_size, hipStream_t stream) {
  (void)in_sizes; (void)n_in; (void)out_size; (void)ws_size;

  const float* hs  = (const float*)d_in[0];  // [1,S,HD]
  const float* pe  = (const float*)d_in[1];  // [POS,HD]
  const float* Wq  = (const float*)d_in[2];
  const float* Wk  = (const float*)d_in[3];
  const float* Wv  = (const float*)d_in[4];
  const float* Wr  = (const float*)d_in[5];
  const float* Wp  = (const float*)d_in[6];
  const float* pds = (const float*)d_in[7];  // [D]
  float* out = (float*)d_out;

  char*  ws  = (char*)d_ws;
  size_t off = 0;
  auto alloc = [&](size_t bytes) -> void* {
    void* p = ws + off;
    off = (off + bytes + 255) & ~(size_t)255;
    return p;
  };

  const size_t xeB = (size_t)SEQ * HID * sizeof(bf16_t);   // 32 MB
  const size_t wB  = (size_t)HID * HID * sizeof(bf16_t);   //  8 MB

  bf16_t* xb   = (bf16_t*)alloc(xeB);  // x bf16; later reused as attn-out
  bf16_t* wqT  = (bf16_t*)alloc(wB);
  bf16_t* wkT  = (bf16_t*)alloc(wB);
  bf16_t* wvT  = (bf16_t*)alloc(wB);
  bf16_t* wrT  = (bf16_t*)alloc(wB);
  bf16_t* wpT  = (bf16_t*)alloc(wB);
  bf16_t* qb   = (bf16_t*)alloc(xeB);
  bf16_t* kb   = (bf16_t*)alloc(xeB);
  bf16_t* vb   = (bf16_t*)alloc(xeB);
  bf16_t* peb  = (bf16_t*)alloc((size_t)128 * HID * sizeof(bf16_t));
  bf16_t* relk = (bf16_t*)alloc((size_t)128 * HID * sizeof(bf16_t));
  float*  qs   = (float*)alloc(DH * sizeof(float));
  // total ~169 MB

  // ---- prep: bf16 conversions, weight transposes, padding, q scales ----
  {
    long nx = (long)SEQ * HID;
    conv_x_kernel<<<dim3((unsigned)((nx + 255) / 256)), 256, 0, stream>>>(hs, xb, nx);
    long nw = (long)HID * HID;
    unsigned gw = (unsigned)((nw + 255) / 256);
    transpose_w_kernel<<<gw, 256, 0, stream>>>(Wq, wqT);
    transpose_w_kernel<<<gw, 256, 0, stream>>>(Wk, wkT);
    transpose_w_kernel<<<gw, 256, 0, stream>>>(Wv, wvT);
    transpose_w_kernel<<<gw, 256, 0, stream>>>(Wr, wrT);
    transpose_w_kernel<<<gw, 256, 0, stream>>>(Wp, wpT);
    posemb_kernel<<<(128 * HID) / 256, 256, 0, stream>>>(pe, peb);
    qscale_kernel<<<1, 128, 0, stream>>>(pds, qs);
  }

  // ---- projections (bf16 WMMA GEMMs) ----
  dim3 gproj(SEQ / 128, HID / 128), bgemm(256);
  gemm_bf16<2><<<gproj, bgemm, 0, stream>>>(xb, wqT, qb, qs, SEQ, HID, HID);
  gemm_bf16<1><<<gproj, bgemm, 0, stream>>>(xb, wkT, kb, nullptr, SEQ, HID, HID);
  gemm_bf16<1><<<gproj, bgemm, 0, stream>>>(xb, wvT, vb, nullptr, SEQ, HID, HID);
  gemm_bf16<1><<<dim3(1, HID / 128), bgemm, 0, stream>>>(peb, wrT, relk, nullptr,
                                                         128, HID, HID);

  // ---- attention core (xb reused as bf16 attn-out; fully overwritten) ----
  attn_kernel<<<dim3(NBLK, HEADS), dim3(32), 0, stream>>>(qb, kb, vb, relk, xb);

  // ---- final projection -> fp32 output ----
  gemm_bf16<0><<<gproj, bgemm, 0, stream>>>(xb, wpT, out, nullptr, SEQ, HID, HID);
}
